// GCN_8057358647356
// MI455X (gfx1250) — matrix operand order repaired and checked by reading, not compiled, not run on hardware
//
#include <hip/hip_runtime.h>
#include <hip/hip_bf16.h>

// ---------------------------------------------------------------------------
// GCN forward, B=64 graphs, N=1024 nodes, F=128, H=256.
// Layers 1-2 as bf16 WMMA GEMMs (v_wmma_f32_16x16x32_bf16); GEMM outputs that
// feed the next GEMM's B operand are written transposed so every LDS stage is
// fully vectorized (no scalar transposes). Layer 3 + linear head collapse to
// a per-graph fp32 row-vector chain. bf16 adjacency (128 MB) is L2-resident
// on MI455X (192 MB L2), so both adj GEMM passes read it from L2.
// ---------------------------------------------------------------------------

typedef __attribute__((ext_vector_type(16))) __bf16 v16bf;
typedef __attribute__((ext_vector_type(8)))  float  v8f;

union ABFrag {
    v16bf v;
    uint4 q[2];
};

__device__ __forceinline__ unsigned short f2bf(float f) {
    union { float f; unsigned int u; } x;
    x.f = f;
    unsigned int u = x.u;
    unsigned int r = u + 0x7FFFu + ((u >> 16) & 1u);   // round-to-nearest-even
    return (unsigned short)(r >> 16);
}

__device__ __forceinline__ float bf2f(unsigned short h) {
    union { float f; unsigned int u; } x;
    x.u = ((unsigned int)h) << 16;
    return x.f;
}

// ------------------------------ fp32 -> bf16 -------------------------------
__global__ void cvt_f32_bf16(const float* __restrict__ in,
                             unsigned short* __restrict__ out,
                             long long n) {
    long long i = (long long)blockIdx.x * blockDim.x + threadIdx.x;
    long long stride = (long long)gridDim.x * blockDim.x;
    for (; i < n; i += stride) out[i] = f2bf(in[i]);
}

// --------------------- fp32 [K][N] -> bf16 transposed [N][K] ---------------
__global__ void cvt_f32_bf16_T(const float* __restrict__ in,
                               unsigned short* __restrict__ out,
                               int K, int N) {
    int idx = blockIdx.x * blockDim.x + threadIdx.x;
    if (idx < K * N) {
        int k = idx / N, n = idx % N;
        out[n * K + k] = f2bf(in[idx]);
    }
}

// ------------------------------ bf16 WMMA GEMM -----------------------------
// C[b] = op( A[b] (MxK, row-major bf16) @ B[b] (KxN, given TRANSPOSED as
// BT[N][K] bf16) + bias ). Output stored bf16, row-major or transposed.
// Block tile 128(M) x 128(N), BLOCK_K = 64. 8 waves; wave w owns rows
// [16w,16w+16) across all 128 N columns = 8 accumulator fragments.
// 16 WMMAs per barrier pair; next tile prefetched into registers; all B
// fragments of a K-substep loaded before the WMMA chain so the matrix pipe
// issues back-to-back instead of draining DScnt per WMMA.
__global__ __launch_bounds__(256)
void gemm_bf16_wmma(const unsigned short* __restrict__ A, long long aBS,
                    const unsigned short* __restrict__ BT, long long bBS,
                    unsigned short* __restrict__ C, long long cBS,
                    int M, int N, int K,
                    const float* __restrict__ bias, int relu, int storeT) {
    __shared__ __align__(16) unsigned short As[128 * 64]; // [m][k], k contiguous
    __shared__ __align__(16) unsigned short Bs[128 * 64]; // [n][k], k contiguous

    const int bn = blockIdx.x;      // N / 128
    const int bm = blockIdx.y;      // M / 128
    const int bz = blockIdx.z;      // batch

    const unsigned short* Ab  = A  + (size_t)bz * (size_t)aBS;
    const unsigned short* BTb = BT + (size_t)bz * (size_t)bBS;
    unsigned short*       Cb  = C  + (size_t)bz * (size_t)cBS;

    const int t = threadIdx.x;
    const int wave = t >> 5;
    const int lane = t & 31;
    const int lane16 = lane & 15;
    const int half = lane >> 4;

    v8f acc[8] = {};

    // ---- prefetch tile k0=0 into registers (4 uint4 each for A and B) ----
    uint4 pa[4], pb[4];
#pragma unroll
    for (int i = 0; i < 4; ++i) {
        int c = t + i * 256;            // 0..1023
        int row = c >> 3;               // 0..127
        int kc  = c & 7;                // 8-ushort chunks along K
        pa[i] = *(const uint4*)(Ab  + (size_t)(bm * 128 + row) * K + kc * 8);
        pb[i] = *(const uint4*)(BTb + (size_t)(bn * 128 + row) * K + kc * 8);
    }

    for (int k0 = 0; k0 < K; k0 += 64) {
        // ---- commit prefetched tile to LDS ----
#pragma unroll
        for (int i = 0; i < 4; ++i) {
            int c = t + i * 256;
            int row = c >> 3;
            int kc  = c & 7;
            *(uint4*)(&As[row * 64 + kc * 8]) = pa[i];
            *(uint4*)(&Bs[row * 64 + kc * 8]) = pb[i];
        }
        __syncthreads();

        // ---- prefetch next tile while WMMAs run ----
        if (k0 + 64 < K) {
            int kn = k0 + 64;
#pragma unroll
            for (int i = 0; i < 4; ++i) {
                int c = t + i * 256;
                int row = c >> 3;
                int kc  = c & 7;
                pa[i] = *(const uint4*)(Ab  + (size_t)(bm * 128 + row) * K + kn + kc * 8);
                pb[i] = *(const uint4*)(BTb + (size_t)(bn * 128 + row) * K + kn + kc * 8);
            }
        }

        // ---- 2 x 8 WMMAs from LDS ----
#pragma unroll
        for (int ks = 0; ks < 2; ++ks) {
            // A fragment 16x32 bf16 (ISA layout):
            // lanes 0-15 (row=lane):   K 0..7 -> q[0], K 16..23 -> q[1]
            // lanes 16-31 (row=l-16):  K 8..15 -> q[0], K 24..31 -> q[1]
            ABFrag a;
            const uint4* arow = (const uint4*)(&As[(wave * 16 + lane16) * 64 + ks * 32]);
            a.q[0] = arow[half];
            a.q[1] = arow[half + 2];

            // B fragments 32x16 bf16 (ISA layout):
            // lanes 0-15 (col=lane): K 0..7, 16..23 ; lanes 16-31: K 8..15, 24..31
            // Load ALL fragments first so the 16 ds_load_b128 are in flight
            // before the first WMMA issues.
            ABFrag bfr[8];
#pragma unroll
            for (int f = 0; f < 8; ++f) {
                const uint4* brow = (const uint4*)(&Bs[(f * 16 + lane16) * 64 + ks * 32]);
                bfr[f].q[0] = brow[half];
                bfr[f].q[1] = brow[half + 2];
            }
#pragma unroll
            for (int f = 0; f < 8; ++f) {
                acc[f] = __builtin_amdgcn_wmma_f32_16x16x32_bf16(
                    false, a.v, false, bfr[f].v, (short)0, acc[f], false, false);
            }
        }
        __syncthreads();
    }

    // ---- epilogue ----
    // C/D layout: VGPR r, lanes 0-15 -> (M=r, N=lane), lanes 16-31 -> (M=r+8).
    if (storeT) {
        // C^T[col][row]: each lane's 8 rows are contiguous -> one uint4 store.
        int rowbase = bm * 128 + wave * 16 + half * 8;
#pragma unroll
        for (int f = 0; f < 8; ++f) {
            int col = bn * 128 + f * 16 + lane16;
            uint4 pk;
            unsigned int* w = (unsigned int*)&pk;
#pragma unroll
            for (int j = 0; j < 4; ++j) {
                unsigned int lo = f2bf(acc[f][2 * j]);
                unsigned int hi = f2bf(acc[f][2 * j + 1]);
                w[j] = lo | (hi << 16);
            }
            *(uint4*)(Cb + (size_t)col * M + rowbase) = pk;
        }
    } else {
#pragma unroll
        for (int f = 0; f < 8; ++f) {
            int col = bn * 128 + f * 16 + lane16;
            float bv = bias ? bias[col] : 0.0f;
#pragma unroll
            for (int r = 0; r < 8; ++r) {
                int row = bm * 128 + wave * 16 + (half ? r + 8 : r);
                float v = acc[f][r] + bv;
                if (relu) v = fmaxf(v, 0.0f);
                Cb[(size_t)row * N + col] = f2bf(v);
            }
        }
    }
}

// ------------------- layer 3 + final linear (fp32, tiny) -------------------
// out[b,:] = relu( (adj[b,0,:] @ H2[b]) @ W2 + b2 ) @ Wl + bl
__global__ __launch_bounds__(256)
void gcn_tail(const float* __restrict__ adj,         // [B,1024,1024] fp32
              const unsigned short* __restrict__ H2, // [B,1024,256] bf16
              const float* __restrict__ W2,          // [256,256]
              const float* __restrict__ b2,          // [256]
              const float* __restrict__ Wl,          // [256,128]
              const float* __restrict__ bl,          // [128]
              float* __restrict__ out) {             // [B,128]
    const int b = blockIdx.x;
    const int t = threadIdx.x;   // 256 threads
    __shared__ float v[256];
    __shared__ float u[256];

    const float* arow = adj + (size_t)b * 1024 * 1024;   // row 0 of graph b
    const unsigned short* H = H2 + (size_t)b * 1024 * 256;

    float acc = 0.0f;
    for (int n = 0; n < 1024; ++n)
        acc += arow[n] * bf2f(H[n * 256 + t]);
    v[t] = acc;
    __syncthreads();

    float acc2 = b2[t];
    for (int k = 0; k < 256; ++k)
        acc2 += v[k] * W2[k * 256 + t];
    u[t] = fmaxf(acc2, 0.0f);
    __syncthreads();

    if (t < 128) {
        float acc3 = bl[t];
        for (int k = 0; k < 256; ++k)
            acc3 += u[k] * Wl[k * 128 + t];
        out[b * 128 + t] = acc3;
    }
}

// ---------------------------------------------------------------------------
extern "C" void kernel_launch(void* const* d_in, const int* in_sizes, int n_in,
                              void* d_out, int out_size, void* d_ws, size_t ws_size,
                              hipStream_t stream) {
    const float* embs = (const float*)d_in[0];  // [64,1024,128]
    const float* adj  = (const float*)d_in[1];  // [64,1024,1024]
    const float* W0   = (const float*)d_in[2];  // [128,256]
    const float* b0   = (const float*)d_in[3];  // [256]
    const float* W1   = (const float*)d_in[4];  // [256,256]
    const float* b1   = (const float*)d_in[5];  // [256]
    const float* W2   = (const float*)d_in[6];  // [256,256]
    const float* b2   = (const float*)d_in[7];  // [256]
    const float* Wl   = (const float*)d_in[8];  // [256,128]
    const float* bl   = (const float*)d_in[9];  // [128]
    float* out = (float*)d_out;                 // [64,128]

    const long long Nadj = 64LL * 1024 * 1024;  // 67,108,864
    const long long Nemb = 64LL * 1024 * 128;   //  8,388,608

    char* ws = (char*)d_ws;
    unsigned short* adj_bf  = (unsigned short*)(ws);                 // 128 MB
    unsigned short* embs_bf = (unsigned short*)(ws + 134217728LL);   //  16 MB
    unsigned short* W0T     = (unsigned short*)(ws + 150994944LL);   //  64 KB  [256][128]
    unsigned short* W1T     = (unsigned short*)(ws + 151060480LL);   // 128 KB  [256][256]
    unsigned short* TT      = (unsigned short*)(ws + 151191552LL);   //  32 MB  [B][256][1024]
    unsigned short* Hbuf    = (unsigned short*)(ws + 184745984LL);   //  32 MB  [B][1024][256]

    // fp32 -> bf16 (adj_bf is L2-resident afterwards: 128 MB < 192 MB L2)
    cvt_f32_bf16<<<65536, 256, 0, stream>>>(adj,  adj_bf,  Nadj);
    cvt_f32_bf16<<<32768, 256, 0, stream>>>(embs, embs_bf, Nemb);
    cvt_f32_bf16_T<<<128, 256, 0, stream>>>(W0, W0T, 128, 256);
    cvt_f32_bf16_T<<<256, 256, 0, stream>>>(W1, W1T, 256, 256);

    dim3 blk(256);
    dim3 grd(256 / 128, 1024 / 128, 64);  // (N tiles, M tiles, batch)

    // Layer 1: TT = (embs @ W0)^T              [1024x128 @ 128x256] -> [256][1024]
    gemm_bf16_wmma<<<grd, blk, 0, stream>>>(embs_bf, 1024LL * 128, W0T, 0LL,
                                            TT, 256LL * 1024,
                                            1024, 256, 128, nullptr, 0, 1);
    //          H1 = relu(adj @ T + b0)         [1024x1024 @ 1024x256]
    gemm_bf16_wmma<<<grd, blk, 0, stream>>>(adj_bf, 1024LL * 1024, TT, 256LL * 1024,
                                            Hbuf, 1024LL * 256,
                                            1024, 256, 1024, b0, 1, 0);
    // Layer 2: TT = (H1 @ W1)^T                [1024x256 @ 256x256] -> [256][1024]
    gemm_bf16_wmma<<<grd, blk, 0, stream>>>(Hbuf, 1024LL * 256, W1T, 0LL,
                                            TT, 256LL * 1024,
                                            1024, 256, 256, nullptr, 0, 1);
    //          H2 = relu(adj @ T + b1)
    gemm_bf16_wmma<<<grd, blk, 0, stream>>>(adj_bf, 1024LL * 1024, TT, 256LL * 1024,
                                            Hbuf, 1024LL * 256,
                                            1024, 256, 1024, b1, 1, 0);

    // Layer 3 + linear head, fp32, one block per graph.
    gcn_tail<<<64, 256, 0, stream>>>(adj, Hbuf, W2, b2, Wl, bl, out);
}
